// ParallelMambaVIX_764504179281
// MI455X (gfx1250) — compile-verified, matrix operand order checked by reading, not run on hardware
//
#include <hip/hip_runtime.h>
#include <hip/hip_bf16.h>
#include <math.h>

// ---------------------------------------------------------------------------
// ParallelMambaVIX forward for MI455X (gfx1250).
//   - Big GEMMs -> v_wmma_f32_16x16x32_f16, f16 operands staged once in ws,
//     16x64 output tile per wave (A fragment reused across 4 accumulators)
//   - Selective scan / conv / LN / softmax / attention -> fp32 VALU kernels
// Sizes: B=8 T=512 BT=4096 D=256 DI=512 DSTATE=16 DTRANK=16 HIDDEN=128
// ---------------------------------------------------------------------------

typedef __attribute__((ext_vector_type(16))) _Float16 v16h;
typedef __attribute__((ext_vector_type(8)))  _Float16 v8h;
typedef __attribute__((ext_vector_type(8)))  float    v8f;

#define BATCH 8
#define TLEN 512
#define BT (BATCH*TLEN)
#define DMODEL 256
#define DINNER 512
#define DSTATE 16
#define DTRANK 16
#define HIDDEN 128

__device__ __forceinline__ float geluf(float x) {
    return 0.5f * x * (1.0f + erff(x * 0.70710678118654752f));
}
__device__ __forceinline__ float sigmf(float x) { return 1.0f / (1.0f + expf(-x)); }

__device__ __forceinline__ float blk_reduce_sum(float v, float* sm, int n) {
    int tid = threadIdx.x;
    sm[tid] = v; __syncthreads();
    for (int s = n >> 1; s > 0; s >>= 1) {
        if (tid < s) sm[tid] += sm[tid + s];
        __syncthreads();
    }
    float r = sm[0]; __syncthreads();
    return r;
}

// ---------------------------------------------------------------------------
// WMMA GEMM:  C[M,N] = A[M,K] @ W[N,K]^T (+ bias[N]),  A/W pre-staged f16.
// One wave computes a 16 x (16*NT) tile; A fragment loaded once per K-step
// and reused for NT accumulators.
// Fragment layout (ISA 7.12.2, 16-bit A 16x32, wave32):
//   lane L: row/col = L%16, hi = L/16; f16 slots are the two contiguous
//   8-half runs at K offsets [kk+8*hi] and [kk+16+8*hi]  (two b128 loads).
// C/D: VGPR r -> M = r + 8*hi, lane%16 -> N.
// ---------------------------------------------------------------------------
template <int NT>
__global__ __launch_bounds__(32)
void wmma_gemm_kernel(const _Float16* __restrict__ A, const _Float16* __restrict__ W,
                      const float* __restrict__ bias, float* __restrict__ C,
                      int M, int N, int K)
{
    const int lane = threadIdx.x;
    const int hi = lane >> 4, lr = lane & 15;
    const int m0 = blockIdx.x * 16, n0 = blockIdx.y * 16 * NT;

    const _Float16* __restrict__ arow = A + (size_t)(m0 + lr) * K + (hi << 3);
    const _Float16* wrow[NT];
    v8f acc[NT];
#pragma unroll
    for (int t = 0; t < NT; ++t) {
        wrow[t] = W + (size_t)(n0 + t * 16 + lr) * K + (hi << 3);
        acc[t] = v8f{};
    }

    for (int kk = 0; kk < K; kk += 32) {
        if (kk + 32 < K)                          // -> global_prefetch_b8
            __builtin_prefetch(arow + kk + 32, 0, 3);
        const v8h al = *(const v8h*)(arow + kk);
        const v8h ah = *(const v8h*)(arow + kk + 16);
        const v16h af = __builtin_shufflevector(al, ah,
            0, 1, 2, 3, 4, 5, 6, 7, 8, 9, 10, 11, 12, 13, 14, 15);
#pragma unroll
        for (int t = 0; t < NT; ++t) {
            const v8h bl = *(const v8h*)(wrow[t] + kk);
            const v8h bh = *(const v8h*)(wrow[t] + kk + 16);
            const v16h bf = __builtin_shufflevector(bl, bh,
                0, 1, 2, 3, 4, 5, 6, 7, 8, 9, 10, 11, 12, 13, 14, 15);
            acc[t] = __builtin_amdgcn_wmma_f32_16x16x32_f16(
                /*neg_a=*/false, af, /*neg_b=*/false, bf,
                /*c_mod=*/(short)0, acc[t], /*reuse_a=*/false, /*reuse_b=*/false);
        }
    }
#pragma unroll
    for (int t = 0; t < NT; ++t) {
        const float bv = bias ? bias[n0 + t * 16 + lr] : 0.0f;
#pragma unroll
        for (int r = 0; r < 8; ++r)
            C[(size_t)(m0 + r + 8 * hi) * N + n0 + t * 16 + lr] = acc[t][r] + bv;
    }
}

// ---------------------- f32 -> f16 staging (weights) ------------------------
__global__ __launch_bounds__(256)
void cvt_f16_kernel(const float* __restrict__ src, _Float16* __restrict__ dst, int n)
{
    int idx = blockIdx.x * 256 + threadIdx.x;
    if (idx < n) dst[idx] = (_Float16)src[idx];
}

// ---------------- embed: h = x @ We^T + be (also f16 copy) ------------------
__global__ __launch_bounds__(256)
void embed_kernel(const float* __restrict__ x, const float* __restrict__ We,
                  const float* __restrict__ be, float* __restrict__ h,
                  _Float16* __restrict__ h16)
{
    int idx = blockIdx.x * 256 + threadIdx.x;       // BT*256
    int bt = idx >> 8, d = idx & 255;
    float acc = be[d];
    const float* xr = x + (size_t)bt * 47;
    const float* wr = We + (size_t)d * 47;
#pragma unroll 1
    for (int k = 0; k < 47; ++k) acc += xr[k] * wr[k];
    h[idx] = acc;
    h16[idx] = (_Float16)acc;
}

// ---------------- FiLM macro MLP: mh[8,256] (single block) ------------------
__global__ __launch_bounds__(256)
void film_mlp_kernel(const float* __restrict__ mc,
                     const float* __restrict__ lng, const float* __restrict__ lnb,
                     const float* __restrict__ W1, const float* __restrict__ b1,
                     const float* __restrict__ W2, const float* __restrict__ b2,
                     float* __restrict__ mh)
{
    __shared__ float mcn[BATCH * 32];
    __shared__ float mh1[BATCH * 256];
    int tid = threadIdx.x;
    {
        int b = tid >> 5, j = tid & 31;
        const float* row = mc + b * 32;
        float mean = 0.f, sq = 0.f;
        for (int k = 0; k < 32; ++k) { mean += row[k]; sq += row[k] * row[k]; }
        mean *= (1.f / 32.f);
        float var = sq * (1.f / 32.f) - mean * mean;
        mcn[tid] = (row[j] - mean) * rsqrtf(var + 1e-5f) * lng[j] + lnb[j];
    }
    __syncthreads();
    for (int b = 0; b < BATCH; ++b) {
        float acc = b1[tid];
        const float* w = W1 + tid * 32;
        for (int k = 0; k < 32; ++k) acc += mcn[b * 32 + k] * w[k];
        mh1[b * 256 + tid] = geluf(acc);
    }
    __syncthreads();
    for (int b = 0; b < BATCH; ++b) {
        float acc = b2[tid];
        const float* w = W2 + tid * 256;
        for (int k = 0; k < 256; ++k) acc += mh1[b * 256 + k] * w[k];
        mh[b * 256 + tid] = acc;
    }
}

// --------- comb16 = f16(mh[b] + time encoding), GEMM operand only ----------
__global__ __launch_bounds__(256)
void comb_kernel(const float* __restrict__ mh, const int* __restrict__ ts,
                 _Float16* __restrict__ comb16)
{
    int idx = blockIdx.x * 256 + threadIdx.x;       // BT*256
    int bt = idx >> 8, d = idx & 255;
    int b = bt >> 9;
    float tod = (float)(ts[bt] % 86400);
    const float c = 9.210340371976184f / 128.f;     // ln(1e4)/half
    float e;
    if (d < 128) e = sinf(tod * expf(-(float)d * c));
    else         e = cosf(tod * expf(-(float)(d - 128) * c));
    comb16[idx] = (_Float16)(mh[b * 256 + d] + e);
}

// ------- causal depthwise conv (K=4) + SiLU: xs (+f16) from xz[...,:512] ----
__global__ __launch_bounds__(256)
void conv_silu_kernel(const float* __restrict__ xz, const float* __restrict__ cw,
                      const float* __restrict__ cb, float* __restrict__ xs,
                      _Float16* __restrict__ xs16)
{
    int idx = blockIdx.x * 256 + threadIdx.x;       // BT*512
    int d = idx & 511, bt = idx >> 9, t = bt & 511;
    float acc = cb[d];
    const float* w = cw + d * 4;
#pragma unroll
    for (int j = 0; j < 4; ++j) {
        int tt = t + j - 3;
        if (tt >= 0) acc += w[j] * xz[(size_t)(bt + j - 3) * 1024 + d];
    }
    float r = acc * sigmf(acc);
    xs[(size_t)bt * 512 + d] = r;
    xs16[(size_t)bt * 512 + d] = (_Float16)r;
}

// ---- selective scan: one thread per (b, d_inner); dt fused in (K=16) -------
// Output y is only ever a GEMM operand -> write f16 directly.
__global__ __launch_bounds__(256)
void scan_kernel(const float* __restrict__ xs, const float* __restrict__ xdbl,
                 const float* __restrict__ xz, const float* __restrict__ Wdt,
                 const float* __restrict__ bdt, const float* __restrict__ Alog,
                 const float* __restrict__ Dp, _Float16* __restrict__ y16)
{
    int idx = blockIdx.x * 256 + threadIdx.x;       // BATCH*DINNER = 4096
    int b = idx >> 9, d = idx & 511;
    float wdt[DSTATE], Av[DSTATE], s[DSTATE];
#pragma unroll
    for (int j = 0; j < DSTATE; ++j) {
        wdt[j] = Wdt[d * DTRANK + j];
        Av[j]  = -expf(Alog[d * DSTATE + j]);
        s[j]   = 0.f;
    }
    const float bd = bdt[d], Dv = Dp[d];
    for (int t = 0; t < TLEN; ++t) {
        const int bt = b * TLEN + t;
        const float* xd = xdbl + (size_t)bt * 48;
        float dtv = bd;
#pragma unroll
        for (int r = 0; r < DTRANK; ++r) dtv += xd[r] * wdt[r];
        dtv = (dtv > 20.f) ? dtv : log1pf(expf(dtv));      // softplus
        const float xv = xs[(size_t)bt * 512 + d];
        float acc = 0.f;
#pragma unroll
        for (int j = 0; j < DSTATE; ++j) {
            s[j] = expf(dtv * Av[j]) * s[j] + dtv * xd[16 + j] * xv;
            acc += xd[32 + j] * s[j];
        }
        const float zv = xz[(size_t)bt * 1024 + 512 + d];
        y16[(size_t)bt * 512 + d] = (_Float16)((acc + xv * Dv) * (zv * sigmf(zv)));
    }
}

// -- h = gamma * LN(h + mo) + beta (gamma|beta packed in gb); also f16 copy --
__global__ __launch_bounds__(256)
void postlayer_kernel(float* __restrict__ h, const float* __restrict__ mo,
                      const float* __restrict__ lng, const float* __restrict__ lnb,
                      const float* __restrict__ gb, _Float16* __restrict__ h16)
{
    __shared__ float red[256];
    int row = blockIdx.x, d = threadIdx.x;
    float v = h[(size_t)row * 256 + d] + mo[(size_t)row * 256 + d];
    float mean = blk_reduce_sum(v, red, 256) * (1.f / 256.f);
    float df = v - mean;
    float var = blk_reduce_sum(df * df, red, 256) * (1.f / 256.f);
    float hn = df * rsqrtf(var + 1e-5f) * lng[d] + lnb[d];
    float r = gb[(size_t)row * 512 + d] * hn + gb[(size_t)row * 512 + 256 + d];
    h[(size_t)row * 256 + d] = r;
    h16[(size_t)row * 256 + d] = (_Float16)r;
}

// ------------------------- attention pooling --------------------------------
__global__ __launch_bounds__(256)
void pool_logits_kernel(const float* __restrict__ h, const float* __restrict__ pw,
                        const float* __restrict__ pb, float* __restrict__ logits)
{
    __shared__ float red[256];
    int row = blockIdx.x, d = threadIdx.x;
    float s = blk_reduce_sum(h[(size_t)row * 256 + d] * pw[d], red, 256);
    if (d == 0) logits[row] = s + pb[0];
}

__global__ __launch_bounds__(256)
void pool_combine_kernel(const float* __restrict__ h, const float* __restrict__ logits,
                         float* __restrict__ pooled)
{
    __shared__ float red[256];
    __shared__ float wexp[TLEN];
    int b = blockIdx.x, tid = threadIdx.x;
    const float* lg = logits + b * TLEN;
    float l0 = lg[tid], l1 = lg[tid + 256];
    red[tid] = fmaxf(l0, l1); __syncthreads();
    for (int s = 128; s > 0; s >>= 1) {
        if (tid < s) red[tid] = fmaxf(red[tid], red[tid + s]);
        __syncthreads();
    }
    float mx = red[0]; __syncthreads();
    float e0 = expf(l0 - mx), e1 = expf(l1 - mx);
    wexp[tid] = e0; wexp[tid + 256] = e1;
    float inv = 1.f / blk_reduce_sum(e0 + e1, red, 256);   // barriers publish wexp
    float acc = 0.f;
    for (int t = 0; t < TLEN; ++t)
        acc += wexp[t] * h[((size_t)b * TLEN + t) * 256 + tid];
    pooled[b * 256 + tid] = acc * inv;
}

// ----------------- cross-attention fusion (B=8, one block) ------------------
__global__ __launch_bounds__(256)
void fuse_kernel(const float* __restrict__ pooled, const float* __restrict__ news,
                 const float* __restrict__ opts,
                 const float* __restrict__ Wq, const float* __restrict__ bq,
                 const float* __restrict__ Wk, const float* __restrict__ bk,
                 const float* __restrict__ Wv, const float* __restrict__ bv,
                 const float* __restrict__ Wo, const float* __restrict__ bo,
                 const float* __restrict__ Wg1, const float* __restrict__ bg1,
                 const float* __restrict__ Wg2, const float* __restrict__ bg2,
                 const float* __restrict__ lng, const float* __restrict__ lnb,
                 float* __restrict__ fused)
{
    __shared__ float sS[256], q[256], k0[256], k1[256], v0[256], v1[256];
    __shared__ float o2[256], g1s[256], pre[256], red[256], aw[8];
    int d = threadIdx.x;
    for (int b = 0; b < BATCH; ++b) {
        sS[d] = pooled[b * 256 + d];
        __syncthreads();
        const float* nrow = news + b * 256;
        const float* orow = opts + b * 256;
        float aq = bq[d], a0 = bk[d], a1 = bk[d], b0 = bv[d], b1v = bv[d];
        const float* wq = Wq + d * 256; const float* wk = Wk + d * 256; const float* wv = Wv + d * 256;
        for (int k = 0; k < 256; ++k) {
            aq += sS[k] * wq[k];
            a0 += nrow[k] * wk[k]; a1 += orow[k] * wk[k];
            b0 += nrow[k] * wv[k]; b1v += orow[k] * wv[k];
        }
        q[d] = aq; k0[d] = a0; k1[d] = a1; v0[d] = b0; v1[d] = b1v;
        __syncthreads();
        red[d] = q[d] * k0[d]; __syncthreads();
        if (d < 4) { float s0 = 0.f; for (int j = 0; j < 64; ++j) s0 += red[d * 64 + j]; aw[d * 2] = s0 * 0.125f; }
        __syncthreads();
        red[d] = q[d] * k1[d]; __syncthreads();
        if (d < 4) {
            float s1 = 0.f; for (int j = 0; j < 64; ++j) s1 += red[d * 64 + j];
            s1 *= 0.125f;
            float s0 = aw[d * 2];
            float mx = fmaxf(s0, s1);
            float e0 = expf(s0 - mx), e1 = expf(s1 - mx), inv = 1.f / (e0 + e1);
            aw[d * 2] = e0 * inv; aw[d * 2 + 1] = e1 * inv;
        }
        __syncthreads();
        int hh = d >> 6;
        red[d] = aw[hh * 2] * v0[d] + aw[hh * 2 + 1] * v1[d];   // attn out
        __syncthreads();
        float ao = bo[d];
        const float* wo = Wo + d * 256;
        for (int k = 0; k < 256; ++k) ao += red[k] * wo[k];
        o2[d] = ao;
        __syncthreads();
        float ag = bg1[d];
        const float* wg = Wg1 + d * 512;
        for (int k = 0; k < 256; ++k) ag += sS[k] * wg[k] + o2[k] * wg[256 + k];
        g1s[d] = geluf(ag);
        __syncthreads();
        float ag2 = bg2[d];
        const float* wg2 = Wg2 + d * 256;
        for (int k = 0; k < 256; ++k) ag2 += g1s[k] * wg2[k];
        float gv = sigmf(ag2);
        pre[d] = gv * o2[d] + (1.f - gv) * sS[d];
        __syncthreads();
        float mean = blk_reduce_sum(pre[d], red, 256) * (1.f / 256.f);
        float df = pre[d] - mean;
        float var = blk_reduce_sum(df * df, red, 256) * (1.f / 256.f);
        fused[b * 256 + d] = df * rsqrtf(var + 1e-5f) * lng[d] + lnb[d];
        __syncthreads();
    }
}

// ------------------------------ output head ---------------------------------
__global__ __launch_bounds__(128)
void head_kernel(const float* __restrict__ fused,
                 const float* __restrict__ W1, const float* __restrict__ b1,
                 const float* __restrict__ g1, const float* __restrict__ bb1,
                 const float* __restrict__ W2, const float* __restrict__ b2,
                 const float* __restrict__ g2, const float* __restrict__ bb2,
                 const float* __restrict__ Wh, const float* __restrict__ bh,
                 float* __restrict__ out)
{
    __shared__ float t1[HIDDEN], t2[HIDDEN], red[HIDDEN];
    int d = threadIdx.x;
    for (int b = 0; b < BATCH; ++b) {
        float acc = b1[d];
        const float* fr = fused + b * 256;
        const float* w = W1 + d * 256;
        for (int k = 0; k < 256; ++k) acc += fr[k] * w[k];
        float mean = blk_reduce_sum(acc, red, HIDDEN) * (1.f / HIDDEN);
        float df = acc - mean;
        float var = blk_reduce_sum(df * df, red, HIDDEN) * (1.f / HIDDEN);
        t1[d] = geluf(df * rsqrtf(var + 1e-5f) * g1[d] + bb1[d]);
        __syncthreads();
        float acc2 = b2[d];
        const float* w2 = W2 + d * HIDDEN;
        for (int k = 0; k < HIDDEN; ++k) acc2 += t1[k] * w2[k];
        mean = blk_reduce_sum(acc2, red, HIDDEN) * (1.f / HIDDEN);
        df = acc2 - mean;
        var = blk_reduce_sum(df * df, red, HIDDEN) * (1.f / HIDDEN);
        t2[d] = geluf(df * rsqrtf(var + 1e-5f) * g2[d] + bb2[d]);
        __syncthreads();
        if (d < 4) {
            float a = bh[d];
            const float* wh = Wh + d * HIDDEN;
            for (int k = 0; k < HIDDEN; ++k) a += t2[k] * wh[k];
            out[b * 4 + d] = a;
        }
        __syncthreads();
    }
}

// ---------------------------------------------------------------------------
extern "C" void kernel_launch(void* const* d_in, const int* in_sizes, int n_in,
                              void* d_out, int out_size, void* d_ws, size_t ws_size,
                              hipStream_t stream)
{
    (void)in_sizes; (void)n_in; (void)out_size; (void)ws_size;
    // Inputs in setup_inputs() insertion order; params flattened recursively
    // in dict/list insertion order.
    const float* x    = (const float*)d_in[0];
    const float* mc   = (const float*)d_in[1];
    const float* news = (const float*)d_in[2];
    const float* opts = (const float*)d_in[3];
    const int*   ts   = (const int*)  d_in[4];
    const float* We   = (const float*)d_in[5];
    const float* be   = (const float*)d_in[6];
    auto LP = [&](int l, int off) { return (const float*)d_in[(l == 0 ? 7 : 20) + off]; };
    const float* f_lng = (const float*)d_in[33];
    const float* f_lnb = (const float*)d_in[34];
    const float* f_W1  = (const float*)d_in[35];
    const float* f_b1  = (const float*)d_in[36];
    const float* f_W2  = (const float*)d_in[37];
    const float* f_b2  = (const float*)d_in[38];
    const float* headsW[2] = { (const float*)d_in[39], (const float*)d_in[40] };
    const float* headsB[2] = { (const float*)d_in[41], (const float*)d_in[42] };
    const float* p_w = (const float*)d_in[43];
    const float* p_b = (const float*)d_in[44];
    const float *Wq=(const float*)d_in[45], *bq=(const float*)d_in[46];
    const float *Wk=(const float*)d_in[47], *bk=(const float*)d_in[48];
    const float *Wv=(const float*)d_in[49], *bv=(const float*)d_in[50];
    const float *Wo=(const float*)d_in[51], *bo=(const float*)d_in[52];
    const float *Wg1=(const float*)d_in[53], *bg1=(const float*)d_in[54];
    const float *Wg2=(const float*)d_in[55], *bg2=(const float*)d_in[56];
    const float *fu_g=(const float*)d_in[57], *fu_b=(const float*)d_in[58];
    const float *h_W1=(const float*)d_in[59], *h_b1=(const float*)d_in[60];
    const float *h_g1=(const float*)d_in[61], *h_bb1=(const float*)d_in[62];
    const float *h_W2=(const float*)d_in[63], *h_b2=(const float*)d_in[64];
    const float *h_g2=(const float*)d_in[65], *h_bb2=(const float*)d_in[66];
    const float *h_Wh=(const float*)d_in[67], *h_bh=(const float*)d_in[68];
    float* out = (float*)d_out;

    // workspace carve-up (units of float, all chunks multiples of 4 floats
    // so every f16 buffer stays 16B-aligned for b128 fragment loads)
    size_t off = 0;
    auto alloc = [&](size_t nfloats) { float* p = (float*)d_ws + off; off += nfloats; return p; };
    float* h      = alloc((size_t)BT * DMODEL);
    float* xz     = alloc((size_t)BT * 2 * DINNER);
    float* xs     = alloc((size_t)BT * DINNER);
    float* xdbl   = alloc((size_t)BT * 48);
    float* mo     = alloc((size_t)BT * DMODEL);
    float* gb     = alloc((size_t)BT * 2 * DMODEL);
    float* mh     = alloc((size_t)BATCH * DMODEL);
    float* logits = alloc(BT);
    float* pooled = alloc((size_t)BATCH * DMODEL);
    float* fusedb = alloc((size_t)BATCH * DMODEL);
    _Float16* h16    = (_Float16*)alloc((size_t)BT * DMODEL / 2);
    _Float16* comb16 = (_Float16*)alloc((size_t)BT * DMODEL / 2);
    _Float16* xs16   = (_Float16*)alloc((size_t)BT * DINNER / 2);
    _Float16* y16    = (_Float16*)alloc((size_t)BT * DINNER / 2);
    _Float16* Win16[2], *Wx16[2], *Wout16[2], *hW16[2];
    for (int l = 0; l < 2; ++l) {
        Win16[l]  = (_Float16*)alloc((size_t)2 * DINNER * DMODEL / 2);   // [1024,256]
        Wx16[l]   = (_Float16*)alloc((size_t)48 * DINNER / 2);           // [48,512]
        Wout16[l] = (_Float16*)alloc((size_t)DMODEL * DINNER / 2);       // [256,512]
        hW16[l]   = (_Float16*)alloc((size_t)2 * DMODEL * DMODEL / 2);   // [512,256]
    }

    // stage all GEMM weights to f16 once per call
    for (int l = 0; l < 2; ++l) {
        cvt_f16_kernel<<<(2 * DINNER * DMODEL + 255) / 256, 256, 0, stream>>>(LP(l, 0), Win16[l],  2 * DINNER * DMODEL);
        cvt_f16_kernel<<<(48 * DINNER + 255) / 256,       256, 0, stream>>>(LP(l, 4), Wx16[l],   48 * DINNER);
        cvt_f16_kernel<<<(DMODEL * DINNER + 255) / 256,   256, 0, stream>>>(LP(l, 9), Wout16[l], DMODEL * DINNER);
        cvt_f16_kernel<<<(2 * DMODEL * DMODEL + 255) / 256, 256, 0, stream>>>(headsW[l], hW16[l], 2 * DMODEL * DMODEL);
    }

    embed_kernel<<<BT * DMODEL / 256, 256, 0, stream>>>(x, We, be, h, h16);
    film_mlp_kernel<<<1, 256, 0, stream>>>(mc, f_lng, f_lnb, f_W1, f_b1, f_W2, f_b2, mh);
    comb_kernel<<<BT * DMODEL / 256, 256, 0, stream>>>(mh, ts, comb16);

    for (int l = 0; l < 2; ++l) {
        const float* b_in  = LP(l, 1);
        const float* cw    = LP(l, 2);  const float* cb    = LP(l, 3);
        const float* W_dt  = LP(l, 5);  const float* b_dt  = LP(l, 6);
        const float* A_log = LP(l, 7);  const float* Dp    = LP(l, 8);
        const float* b_out = LP(l, 10);
        const float* ln_g  = LP(l, 11); const float* ln_b  = LP(l, 12);

        wmma_gemm_kernel<4><<<dim3(BT / 16, (2 * DINNER) / 64), 32, 0, stream>>>(
            h16, Win16[l], b_in, xz, BT, 2 * DINNER, DMODEL);
        conv_silu_kernel<<<BT * DINNER / 256, 256, 0, stream>>>(xz, cw, cb, xs, xs16);
        wmma_gemm_kernel<3><<<dim3(BT / 16, 1), 32, 0, stream>>>(
            xs16, Wx16[l], nullptr, xdbl, BT, 48, DINNER);
        scan_kernel<<<BATCH * DINNER / 256, 256, 0, stream>>>(
            xs, xdbl, xz, W_dt, b_dt, A_log, Dp, y16);
        wmma_gemm_kernel<4><<<dim3(BT / 16, DMODEL / 64), 32, 0, stream>>>(
            y16, Wout16[l], b_out, mo, BT, DMODEL, DINNER);
        wmma_gemm_kernel<4><<<dim3(BT / 16, (2 * DMODEL) / 64), 32, 0, stream>>>(
            comb16, hW16[l], headsB[l], gb, BT, 2 * DMODEL, DMODEL);
        postlayer_kernel<<<BT, 256, 0, stream>>>(h, mo, ln_g, ln_b, gb, h16);
    }

    pool_logits_kernel<<<BT, 256, 0, stream>>>(h, p_w, p_b, logits);
    pool_combine_kernel<<<BATCH, 256, 0, stream>>>(h, logits, pooled);
    fuse_kernel<<<1, 256, 0, stream>>>(pooled, news, opts, Wq, bq, Wk, bk, Wv, bv,
                                       Wo, bo, Wg1, bg1, Wg2, bg2, fu_g, fu_b, fusedb);
    head_kernel<<<1, 128, 0, stream>>>(fusedb, h_W1, h_b1, h_g1, h_bb1,
                                       h_W2, h_b2, h_g2, h_bb2, h_Wh, h_bh, out);
}
